// PolymorphicAttention_59158879535177
// MI455X (gfx1250) — matrix operand order repaired and checked by reading, not compiled, not run on hardware
//
#include <hip/hip_runtime.h>
#include <hip/hip_bf16.h>

// ---------------------------------------------------------------------------
// PolymorphicAttention for MI455X (gfx1250, wave32, WMMA 16x16x32 f16/f32-acc)
//
// Pipeline:
//   k0: x (f32) -> Xh (f16)
//   k1: WMMA GEMM  Xh[16384x1024] * Wqkv[1024x3072] + b  -> scatter Q/K/V f16 [B,H,S,64]
//   k2: row norms  q_sq, k_sq (f32) [B,H,S]
//   k3: flash attention with distance warp + online softmax -> Oh f16 [B,S,1024]
//       (K tiles staged via GLOBAL_LOAD_ASYNC_TO_LDS_B128, ASYNCcnt-fenced)
//   k4: WMMA GEMM  Oh * Wout[1024x1024] + b_out -> out f32
// ---------------------------------------------------------------------------

typedef __attribute__((ext_vector_type(16))) _Float16 v16h;
typedef __attribute__((ext_vector_type(8)))  float    v8f;

#define BATCH   4
#define SEQ     4096
#define DMODEL  1024
#define NHEADS  16
#define HDIM    64
#define MROWS   (BATCH * SEQ)          // 16384
#define BHROWS  (BATCH * NHEADS)       // 64

// workspace layout (bytes)
#define OFF_XH   ((size_t)0)                         // 16384*1024 f16 = 32MB
#define OFF_QH   ((size_t)33554432)                  // 32MB
#define OFF_KH   ((size_t)67108864)                  // 32MB
#define OFF_VH   ((size_t)100663296)                 // 32MB
#define OFF_OH   ((size_t)134217728)                 // 32MB
#define OFF_QSQ  ((size_t)167772160)                 // 1MB  (f32 [BH*S])
#define OFF_KSQ  ((size_t)168820736)                 // 1MB

__device__ __forceinline__ v8f wmma32(v16h a, v16h b, v8f c) {
  return __builtin_amdgcn_wmma_f32_16x16x32_f16(
      /*neg_a=*/false, a, /*neg_b=*/false, b,
      /*c_mod=*/(short)0, c, /*reuse_a=*/false, /*reuse_b=*/false);
}

// A-matrix fragment (16x32 f16, MxK). Contiguous-K layout (same as frag_b):
//   lanes 0-15 : M=lane,    elems i -> K=i
//   lanes16-31 : M=lane-16, elems i -> K=16+i
// `p` points at the tile's row 0; rows are `ld` halfs apart, K contiguous.
__device__ __forceinline__ v16h frag_a(const _Float16* p, int ld, int koff) {
  int lane = threadIdx.x & 31;
  const _Float16* r = p + (size_t)(lane & 15) * ld + koff + ((lane < 16) ? 0 : 16);
  v16h f;
#pragma unroll
  for (int i = 0; i < 8; ++i) f[i] = r[i];
#pragma unroll
  for (int i = 0; i < 8; ++i) f[8 + i] = r[8 + i];
  return f;
}

// B-matrix fragment (32x16 f16, KxN), stored N-major in LDS (row n, K contiguous).
//   lanes 0-15 : N=lane,    elems i -> K=i
//   lanes16-31 : N=lane-16, elems i -> K=16+i
__device__ __forceinline__ v16h frag_b(const _Float16* p, int ld, int koff) {
  int lane = threadIdx.x & 31;
  const _Float16* r = p + (size_t)(lane & 15) * ld + koff + ((lane < 16) ? 0 : 16);
  v16h f;
#pragma unroll
  for (int i = 0; i < 16; ++i) f[i] = r[i];
  return f;
}

// ---------------------------------------------------------------- k0: convert
__global__ void cvt_f32_to_f16(const float* __restrict__ src,
                               _Float16* __restrict__ dst, int n) {
  int i = (blockIdx.x * blockDim.x + threadIdx.x) * 4;
  if (i + 3 < n) {
    float4 v = *(const float4*)(src + i);
    dst[i + 0] = (_Float16)v.x;
    dst[i + 1] = (_Float16)v.y;
    dst[i + 2] = (_Float16)v.z;
    dst[i + 3] = (_Float16)v.w;
  }
}

// ---------------------------------------------------- k1/k4: tiled WMMA GEMM
// C[M x N] = A_f16[M x 1024] * B_f32[1024 x N] + bias
// Block tile 128x128, K-chunk 32, 256 threads = 8 waves (2 x 4), wave tile 64x32.
// MODE 0: scatter f16 into Q/K/V [BH][S][64].  MODE 1: f32 store to Cout[M x N].
template <int MODE>
__global__ __launch_bounds__(256) void gemm_wmma_kernel(
    const _Float16* __restrict__ A, const float* __restrict__ B,
    const float* __restrict__ bias, int N,
    _Float16* __restrict__ Qh, _Float16* __restrict__ Kh,
    _Float16* __restrict__ Vh, float* __restrict__ Cout) {
  const int LDA = 48, LDB = 48;                       // halfs; 96B rows (16B aligned)
  __shared__ _Float16 Asm[128 * 48];
  __shared__ _Float16 Bsm[128 * 48];

  int tid  = threadIdx.x;
  int wave = tid >> 5, lane = tid & 31;
  int wm = wave >> 2, wn = wave & 3;                  // 2 x 4 wave grid
  int m0 = blockIdx.y * 128, n0 = blockIdx.x * 128;

  v8f acc[4][2] = {};

  for (int k0 = 0; k0 < DMODEL; k0 += 32) {
    // A tile: 128 x 32 f16 (4096 elems = 512 x 8-half groups)
    for (int t = tid; t < 512; t += 256) {
      int row = t >> 2, c8 = (t & 3) * 8;
      const _Float16* src = A + (size_t)(m0 + row) * DMODEL + k0 + c8;
      *(uint4*)(Asm + row * LDA + c8) = *(const uint4*)src;
      if (k0 + 32 < DMODEL) __builtin_prefetch(src + 32, 0, 3);
    }
    // B tile: 32 x 128 f32 -> Bsm[n][k] (transposed, f16)
    for (int t = tid; t < 1024; t += 256) {
      int kr = t >> 5, nq = (t & 31) * 4;
      const float* src = B + (size_t)(k0 + kr) * N + n0 + nq;
      float4 v = *(const float4*)src;
      Bsm[(nq + 0) * LDB + kr] = (_Float16)v.x;
      Bsm[(nq + 1) * LDB + kr] = (_Float16)v.y;
      Bsm[(nq + 2) * LDB + kr] = (_Float16)v.z;
      Bsm[(nq + 3) * LDB + kr] = (_Float16)v.w;
    }
    __syncthreads();

    v16h af[4];
#pragma unroll
    for (int mt = 0; mt < 4; ++mt)
      af[mt] = frag_a(Asm + (wm * 64 + mt * 16) * LDA, LDA, 0);
#pragma unroll
    for (int nt = 0; nt < 2; ++nt) {
      v16h bf = frag_b(Bsm + (wn * 32 + nt * 16) * LDB, LDB, 0);
#pragma unroll
      for (int mt = 0; mt < 4; ++mt) acc[mt][nt] = wmma32(af[mt], bf, acc[mt][nt]);
    }
    __syncthreads();
  }

  int rhalf = (lane >= 16) ? 8 : 0;
#pragma unroll
  for (int mt = 0; mt < 4; ++mt)
#pragma unroll
    for (int nt = 0; nt < 2; ++nt)
#pragma unroll
      for (int e = 0; e < 8; ++e) {
        int m = m0 + wm * 64 + mt * 16 + e + rhalf;
        int n = n0 + wn * 32 + nt * 16 + (lane & 15);
        float v = acc[mt][nt][e] + bias[n];
        if (MODE == 0) {
          int which = n >> 10;          // 0=q 1=k 2=v
          int dcol  = n & 1023;
          int h = dcol >> 6, hd = dcol & 63;
          int b = m >> 12, s = m & 4095;
          _Float16* dst = (which == 0) ? Qh : ((which == 1) ? Kh : Vh);
          dst[(((size_t)(b * NHEADS + h)) * SEQ + s) * HDIM + hd] = (_Float16)v;
        } else {
          Cout[(size_t)m * N + n] = v;
        }
      }
}

// ----------------------------------------------------------------- k2: norms
__global__ void row_norms_kernel(const _Float16* __restrict__ X,
                                 float* __restrict__ nsq, int nrows) {
  int r = blockIdx.x * blockDim.x + threadIdx.x;
  if (r >= nrows) return;
  const _Float16* p = X + (size_t)r * HDIM;
  float s = 0.f;
#pragma unroll 8
  for (int i = 0; i < HDIM; ++i) {
    float v = (float)p[i];
    s += v * v;
  }
  nsq[r] = s;
}

// -------------------------------------------------------- k3: flash attention
// grid (BH, S/128). 8 waves; wave w owns 16 query rows. K tiles staged with
// async global->LDS DMA (ASYNCcnt), V transposed through VGPRs; online softmax;
// distance warp done in f32 VALU.
__global__ __launch_bounds__(256) void attn_kernel(
    const _Float16* __restrict__ Qh, const _Float16* __restrict__ Kh,
    const _Float16* __restrict__ Vh, const float* __restrict__ qsq,
    const float* __restrict__ ksq, _Float16* __restrict__ Oh) {
  const int LDK = 72;                                 // 144B rows, 16B aligned
  __shared__ _Float16 Ksm[64 * 72];                   // [key][hd]
  __shared__ _Float16 Vsm[64 * 72];                   // [hd][key] (transposed)
  __shared__ _Float16 Psm[8 * 16 * 72];               // per-wave P staging
  __shared__ float ksq_s[64];
  __shared__ float mk_s[64];

  int bh  = blockIdx.x;
  int qt  = blockIdx.y;
  int tid = threadIdx.x;
  int wave = tid >> 5, lane = tid & 31;
  int rhalf = (lane >= 16) ? 8 : 0;
  const size_t base = (size_t)bh * SEQ * HDIM;

  int qrow0 = qt * 128 + wave * 16;

  // resident Q fragments (A matrix 16 x 64 -> two 16x32 frags), direct from global
  v16h qf0 = frag_a(Qh + base + (size_t)qrow0 * HDIM, HDIM, 0);
  v16h qf1 = frag_a(Qh + base + (size_t)qrow0 * HDIM, HDIM, 32);

  float qs[8], mq[8], mrow[8], lrow[8];
#pragma unroll
  for (int e = 0; e < 8; ++e) {
    int r = qrow0 + e + rhalf;
    qs[e]   = qsq[(size_t)bh * SEQ + r];
    mq[e]   = sqrtf(qs[e]);
    mrow[e] = -INFINITY;
    lrow[e] = 0.f;
  }
  v8f o[4] = {};

#pragma unroll 1
  for (int j = 0; j < SEQ / 64; ++j) {
    int k0 = j * 64;
    // K tile: [key][hd], lane-wise async DMA global -> LDS (16B per op).
    // LDS operand = low 32 bits of the generic shared address (ISA: flat->LDS
    // aperture keeps addr[31:0]); tracked by ASYNCcnt, fenced before barrier.
    for (int t = tid; t < 512; t += 256) {
      int row = t >> 3, c8 = (t & 7) * 8;
      const _Float16* gsrc = Kh + base + (size_t)(k0 + row) * HDIM + c8;
      unsigned lds_off = (unsigned)(uintptr_t)(Ksm + row * LDK + c8);
      asm volatile("global_load_async_to_lds_b128 %0, %1, off"
                   :
                   : "v"(lds_off), "v"((unsigned long long)(uintptr_t)gsrc)
                   : "memory");
    }
    // V tile transposed: Vsm[hd][key] (needs shuffle, stays on VGPR path)
    for (int t = tid; t < 512; t += 256) {
      int row = t >> 3, c8 = (t & 7) * 8;
      const _Float16* src = Vh + base + (size_t)(k0 + row) * HDIM + c8;
#pragma unroll
      for (int u = 0; u < 8; ++u) Vsm[(c8 + u) * LDK + row] = src[u];
    }
    if (tid < 64) {
      float kv = ksq[(size_t)bh * SEQ + k0 + tid];
      ksq_s[tid] = kv;
      mk_s[tid]  = sqrtf(kv);
    }
    asm volatile("s_wait_asynccnt 0x0" ::: "memory");
    __syncthreads();

    // S = Q * K^T : per N-tile (16 keys) two WMMAs over hd halves
    v8f sc[4];
#pragma unroll
    for (int nt = 0; nt < 4; ++nt) {
      v16h kf0 = frag_b(Ksm + nt * 16 * LDK, LDK, 0);
      v16h kf1 = frag_b(Ksm + nt * 16 * LDK, LDK, 32);
      v8f c = {};
      c = wmma32(qf0, kf0, c);
      c = wmma32(qf1, kf1, c);
      sc[nt] = c;
    }

    // distance warp + force, track new row max
    float fv[4][8], nmax[8];
#pragma unroll
    for (int e = 0; e < 8; ++e) nmax[e] = mrow[e];
#pragma unroll
    for (int nt = 0; nt < 4; ++nt) {
      int col = nt * 16 + (lane & 15);
#pragma unroll
      for (int e = 0; e < 8; ++e) {
        float qk = sc[nt][e];
        float d2 = fmaxf(qs[e] + ksq_s[col] - 2.f * qk, 0.f);
        d2 = d2 * (1.f + 0.15f * __cosf(sqrtf(d2 + 1e-6f)));
        float f = (mq[e] * mk_s[col]) / (d2 + 1e-6f);
        fv[nt][e] = f;
        nmax[e] = fmaxf(nmax[e], f);
      }
    }
    // row max across the 16 lanes of each half (wave32 butterfly)
#pragma unroll
    for (int msk = 1; msk < 16; msk <<= 1)
#pragma unroll
      for (int e = 0; e < 8; ++e)
        nmax[e] = fmaxf(nmax[e], __shfl_xor(nmax[e], msk, 32));

    float psum[8];
#pragma unroll
    for (int e = 0; e < 8; ++e) {
      float alpha = __expf(mrow[e] - nmax[e]);        // 0 on first tile
      lrow[e] *= alpha;
#pragma unroll
      for (int nt = 0; nt < 4; ++nt) o[nt][e] *= alpha;
      mrow[e] = nmax[e];
      psum[e] = 0.f;
    }
    // P = exp(f - m), stage into LDS to re-fragment as an A matrix
    _Float16* Pw = Psm + wave * 16 * LDK;
#pragma unroll
    for (int nt = 0; nt < 4; ++nt) {
      int col = nt * 16 + (lane & 15);
#pragma unroll
      for (int e = 0; e < 8; ++e) {
        float p = __expf(fv[nt][e] - mrow[e]);
        psum[e] += p;
        Pw[(e + rhalf) * LDK + col] = (_Float16)p;
      }
    }
#pragma unroll
    for (int msk = 1; msk < 16; msk <<= 1)
#pragma unroll
      for (int e = 0; e < 8; ++e) psum[e] += __shfl_xor(psum[e], msk, 32);
#pragma unroll
    for (int e = 0; e < 8; ++e) lrow[e] += psum[e];

    // O += P * V   (A = P 16x64, B = V 64x16-per-tile from transposed Vsm)
    v16h pf0 = frag_a(Pw, LDK, 0);
    v16h pf1 = frag_a(Pw, LDK, 32);
#pragma unroll
    for (int nt = 0; nt < 4; ++nt) {
      v16h vf0 = frag_b(Vsm + nt * 16 * LDK, LDK, 0);
      v16h vf1 = frag_b(Vsm + nt * 16 * LDK, LDK, 32);
      o[nt] = wmma32(pf0, vf0, o[nt]);
      o[nt] = wmma32(pf1, vf1, o[nt]);
    }
    __syncthreads();
  }

  // epilogue: normalize and write Oh [B,S,DMODEL]
  int b = bh >> 4, h = bh & 15;
#pragma unroll
  for (int nt = 0; nt < 4; ++nt) {
    int hd = nt * 16 + (lane & 15);
#pragma unroll
    for (int e = 0; e < 8; ++e) {
      int srow = qrow0 + e + rhalf;
      float val = o[nt][e] / lrow[e];
      Oh[((size_t)(b * SEQ + srow)) * DMODEL + h * HDIM + hd] = (_Float16)val;
    }
  }
}

// ---------------------------------------------------------------------------
extern "C" void kernel_launch(void* const* d_in, const int* in_sizes, int n_in,
                              void* d_out, int out_size, void* d_ws, size_t ws_size,
                              hipStream_t stream) {
  const float* x    = (const float*)d_in[0];
  const float* Wqkv = (const float*)d_in[1];
  const float* bqkv = (const float*)d_in[2];
  const float* Wout = (const float*)d_in[3];
  const float* bout = (const float*)d_in[4];
  float* out = (float*)d_out;

  char* ws = (char*)d_ws;
  _Float16* Xh  = (_Float16*)(ws + OFF_XH);
  _Float16* Qh  = (_Float16*)(ws + OFF_QH);
  _Float16* Kh  = (_Float16*)(ws + OFF_KH);
  _Float16* Vh  = (_Float16*)(ws + OFF_VH);
  _Float16* Oh  = (_Float16*)(ws + OFF_OH);
  float*    qsq = (float*)(ws + OFF_QSQ);
  float*    ksq = (float*)(ws + OFF_KSQ);

  // k0: x -> f16
  {
    int n = MROWS * DMODEL;                           // 16,777,216
    cvt_f32_to_f16<<<n / (256 * 4), 256, 0, stream>>>(x, Xh, n);
  }
  // k1: QKV projection GEMM (N = 3072), scatter Q/K/V head-major
  gemm_wmma_kernel<0><<<dim3(3 * DMODEL / 128, MROWS / 128), 256, 0, stream>>>(
      Xh, Wqkv, bqkv, 3 * DMODEL, Qh, Kh, Vh, nullptr);
  // k2: row norms
  {
    int nrows = BHROWS * SEQ;                         // 262,144
    row_norms_kernel<<<nrows / 256, 256, 0, stream>>>(Qh, qsq, nrows);
    row_norms_kernel<<<nrows / 256, 256, 0, stream>>>(Kh, ksq, nrows);
  }
  // k3: attention
  attn_kernel<<<dim3(BHROWS, SEQ / 128), 256, 0, stream>>>(Qh, Kh, Vh, qsq, ksq, Oh);
  // k4: output projection GEMM (N = 1024), f32 out
  gemm_wmma_kernel<1><<<dim3(DMODEL / 128, MROWS / 128), 256, 0, stream>>>(
      Oh, Wout, bout, DMODEL, nullptr, nullptr, nullptr, out);
}
